// VectorQuantizer_73710228734428
// MI455X (gfx1250) — matrix operand order repaired and checked by reading, not compiled, not run on hardware
//
#include <hip/hip_runtime.h>
#include <hip/hip_bf16.h>

typedef __attribute__((ext_vector_type(16))) __bf16 v16bf;
typedef __attribute__((ext_vector_type(8)))  __bf16 bf16x8;
typedef __attribute__((ext_vector_type(4)))  __bf16 bf16x4;
typedef __attribute__((ext_vector_type(8)))  float  v8f;

static constexpr int D      = 768;
static constexpr int K      = 8192;
static constexpr int NROWS  = 16384;      // 8 * 2048
static constexpr int SP     = 776;        // padded LDS stride (bf16 elems): conflict-free b128
static constexpr int MT     = 64;         // rows per workgroup (4 M-subtiles)
static constexpr int NT     = 128;        // codes per outer iteration (8 waves x 16)
static constexpr int NCH    = D / 32;     // 24 K-chunks per dot product
static constexpr int QN     = NROWS * D;  // quantized element count

#define SHUF16(lo, hi) __builtin_shufflevector(lo, hi, 0,1,2,3,4,5,6,7,8,9,10,11,12,13,14,15)

// ---------------- Kernel A: codebook -> bf16, per-code squared norms ----------------
__global__ void __launch_bounds__(256) k_prep(const float* __restrict__ emb,
                                              __bf16* __restrict__ embB,
                                              float* __restrict__ eNorm) {
    const int k = blockIdx.x;
    const int t = threadIdx.x;
    const float* row = emb + (size_t)k * D;
    float ss = 0.0f;
    for (int d = t; d < D; d += 256) {
        float v = row[d];
        ss += v * v;
        embB[(size_t)k * D + d] = (__bf16)v;
    }
    __shared__ float red[256];
    red[t] = ss;
    __syncthreads();
    for (int s = 128; s > 0; s >>= 1) {
        if (t < s) red[t] += red[t + s];
        __syncthreads();
    }
    if (t == 0) eNorm[k] = red[0];
}

// ---------------- Kernel B: WMMA distance GEMM + running argmin ----------------
// score(row, code) = ||e||^2 - 2 * <x, e>   (row norm dropped: constant per row)
// 8 waves; wave w owns code-subtile n_sub = w (16 codes). Each wave computes all FOUR
// 16-row M-subtiles against its code tile -> each B fragment (global, L2-resident)
// feeds 4 WMMAs. A stays in LDS: an asm compiler fence per c0 iteration defeats LICM
// so the 768-VGPR A-hoist (which spilled to scratch last round) cannot happen; A is
// re-read via cheap conflict-free ds_load_b128 instead. B is double-buffered.
__global__ void __launch_bounds__(256) k_argmin(const float* __restrict__ z,
                                                const __bf16* __restrict__ embB,
                                                const float* __restrict__ eNorm,
                                                int* __restrict__ outIdx) {
    __shared__ __bf16 ldsA[MT * SP];     // 99,328 B
    __shared__ float  sV[MT * 8];
    __shared__ int    sI[MT * 8];

    const int t = threadIdx.x;
    const int rowBase = blockIdx.x * MT;

    // Stage 64 rows of z_e into LDS as bf16 (vectorized: float4 -> bf16x4)
    for (int i = t; i < MT * (D / 4); i += 256) {
        int m  = i / (D / 4);
        int d4 = i - m * (D / 4);
        float4 v = *(const float4*)(z + (size_t)(rowBase + m) * D + d4 * 4);
        bf16x4 o = {(__bf16)v.x, (__bf16)v.y, (__bf16)v.z, (__bf16)v.w};
        *(bf16x4*)(ldsA + m * SP + d4 * 4) = o;
    }
    __syncthreads();

    const int lane  = t & 31;
    const int n_sub = t >> 5;      // 0..7 -> codes [c0 + n_sub*16, +16)
    const int col   = lane & 15;
    const int hi    = lane >> 4;   // 0 or 1 (lane half)

    // A fragment bases per M-subtile: 16-bit A 16x32 layout — lanes 0-15: M=lane,
    // K groups {0..7,16..23}; lanes 16-31: same M, K groups {8..15,24..31}
    const __bf16* aptr[4];
#pragma unroll
    for (int s = 0; s < 4; ++s) aptr[s] = ldsA + (s * 16 + col) * SP + hi * 8;

    float minv[4][8];
    int   mini[4][8];
#pragma unroll
    for (int s = 0; s < 4; ++s)
#pragma unroll
        for (int r = 0; r < 8; ++r) { minv[s][r] = 3.402823466e38f; mini[s][r] = 0; }

    for (int c0 = 0; c0 < K; c0 += NT) {
        // Fence: forbid hoisting the (loop-invariant) LDS A-fragment loads out of
        // this loop — that transform needs ~768 VGPRs and spills to scratch.
        asm volatile("" ::: "memory");

        const int code = c0 + n_sub * 16 + col;
        // B fragment: 16-bit B 32x16 — lane = column (code), lanes 0-15 K={0..7,16..23}, 16-31 K={8..15,24..31}
        const __bf16* bptr = embB + (size_t)code * D + hi * 8;
        const float en = eNorm[code];   // issue early; consumed after the GEMM loop

        v8f acc[4];
#pragma unroll
        for (int s = 0; s < 4; ++s) acc[s] = (v8f){0.f,0.f,0.f,0.f,0.f,0.f,0.f,0.f};

        // 2-deep double-buffered B stream (parity buffers, fully unrolled)
        bf16x8 bl[2], bh[2];
        bl[0] = *(const bf16x8*)(bptr);
        bh[0] = *(const bf16x8*)(bptr + 16);
#pragma unroll
        for (int c = 0; c < NCH; ++c) {
            const int d = c * 32;
            if (c + 1 < NCH) {
                bl[(c + 1) & 1] = *(const bf16x8*)(bptr + d + 32);
                bh[(c + 1) & 1] = *(const bf16x8*)(bptr + d + 48);
            }
            v16bf b = SHUF16(bl[c & 1], bh[c & 1]);
#pragma unroll
            for (int s = 0; s < 4; ++s) {
                bf16x8 alo = *(const bf16x8*)(aptr[s] + d);
                bf16x8 ahi = *(const bf16x8*)(aptr[s] + d + 16);
                v16bf a = SHUF16(alo, ahi);
                acc[s] = __builtin_amdgcn_wmma_f32_16x16x32_bf16(
                    false, a, false, b, (short)0, acc[s], false, false);
            }
        }

#pragma unroll
        for (int s = 0; s < 4; ++s) {
#pragma unroll
            for (int r = 0; r < 8; ++r) {
                float sc = en - 2.0f * acc[s][r];
                if (sc < minv[s][r]) { minv[s][r] = sc; mini[s][r] = code; }  // strict <: first index
            }
        }
    }

    // Reduce across the 16 columns of each lane-half (xor masks stay within the half)
#pragma unroll
    for (int s = 0; s < 4; ++s) {
#pragma unroll
        for (int r = 0; r < 8; ++r) {
#pragma unroll
            for (int m = 1; m < 16; m <<= 1) {
                float ov = __shfl_xor(minv[s][r], m, 32);
                int   oi = __shfl_xor(mini[s][r], m, 32);
                if (ov < minv[s][r] || (ov == minv[s][r] && oi < mini[s][r])) {
                    minv[s][r] = ov; mini[s][r] = oi;
                }
            }
        }
    }

    // C layout: VGPR r holds M = r (lanes 0-15) / r+8 (lanes 16-31)
    if (col == 0) {
#pragma unroll
        for (int s = 0; s < 4; ++s) {
#pragma unroll
            for (int r = 0; r < 8; ++r) {
                int m = s * 16 + hi * 8 + r;
                sV[m * 8 + n_sub] = minv[s][r];
                sI[m * 8 + n_sub] = mini[s][r];
            }
        }
    }
    __syncthreads();

    if (t < MT) {
        float bv = sV[t * 8];
        int   bi = sI[t * 8];
#pragma unroll
        for (int j = 1; j < 8; ++j) {
            float v2 = sV[t * 8 + j];
            int   i2 = sI[t * 8 + j];
            if (v2 < bv || (v2 == bv && i2 < bi)) { bv = v2; bi = i2; }
        }
        outIdx[rowBase + t] = bi;
    }
}

// ---------------- Kernel C: exact fp32 gather + per-row loss partials ----------------
__global__ void __launch_bounds__(256) k_gather(const float* __restrict__ z,
                                                const float* __restrict__ emb,
                                                const int* __restrict__ idx,
                                                float* __restrict__ outQ,
                                                float* __restrict__ outIdxF,
                                                float* __restrict__ part) {
    const int r = blockIdx.x;
    const int t = threadIdx.x;
    const int id = idx[r];
    const float* e  = emb + (size_t)id * D;
    const float* zr = z   + (size_t)r  * D;
    float*       o  = outQ + (size_t)r * D;
    float ss = 0.0f;
    for (int d = t; d < D; d += 256) {
        float ev = e[d];
        float zv = zr[d];
        o[d] = ev;                         // quantized_st == quantized numerically
        float df = ev - zv;
        ss += df * df;
    }
    __shared__ float red[256];
    red[t] = ss;
    __syncthreads();
    for (int s = 128; s > 0; s >>= 1) {
        if (t < s) red[t] += red[t + s];
        __syncthreads();
    }
    if (t == 0) {
        part[r] = red[0];
        outIdxF[r] = (float)id;
    }
}

// ---------------- Kernel D: deterministic fixed-order loss reduction ----------------
__global__ void __launch_bounds__(256) k_loss(const float* __restrict__ part,
                                              float* __restrict__ lossOut) {
    const int t = threadIdx.x;
    float s = 0.0f;
    for (int i = t; i < NROWS; i += 256) s += part[i];
    __shared__ float red[256];
    red[t] = s;
    __syncthreads();
    for (int k = 128; k > 0; k >>= 1) {
        if (t < k) red[t] += red[t + k];
        __syncthreads();
    }
    if (t == 0) lossOut[0] = 0.25f * red[0] / (float)QN;
}

extern "C" void kernel_launch(void* const* d_in, const int* in_sizes, int n_in,
                              void* d_out, int out_size, void* d_ws, size_t ws_size,
                              hipStream_t stream) {
    const float* z_e = (const float*)d_in[0];     // [16384, 768]
    const float* emb = (const float*)d_in[1];     // [8192, 768]

    // Workspace layout (bytes)
    char* ws = (char*)d_ws;
    __bf16* embB  = (__bf16*)ws;                                   // 12,582,912 B
    float*  eNorm = (float*)(ws + (size_t)K * D * 2);              //     32,768 B
    int*    wsIdx = (int*)(ws + (size_t)K * D * 2 + K * 4);        //     65,536 B
    float*  wsPart= (float*)(ws + (size_t)K * D * 2 + K * 4 + NROWS * 4); // 65,536 B

    float* outQ    = (float*)d_out;
    float* outIdxF = outQ + QN;
    float* outLoss = outQ + QN + NROWS;

    k_prep  <<<K,         256, 0, stream>>>(emb, embB, eNorm);
    k_argmin<<<NROWS/MT,  256, 0, stream>>>(z_e, embB, eNorm, wsIdx);
    k_gather<<<NROWS,     256, 0, stream>>>(z_e, emb, wsIdx, outQ, outIdxF, wsPart);
    k_loss  <<<1,         256, 0, stream>>>(wsPart, outLoss);
}
